// Attention_32435593019960
// MI455X (gfx1250) — matrix operand order repaired and checked by reading, not compiled, not run on hardware
//
#include <hip/hip_runtime.h>
#include <hip/hip_bf16.h>
#include <cmath>

#define DIM      1024
#define HEADS    16
#define DIM_HEAD 64
#define INNER    (HEADS * DIM_HEAD)
#define LN_EPS   1e-5f

typedef __attribute__((ext_vector_type(16))) _Float16 v16h;
typedef __attribute__((ext_vector_type(8)))  _Float16 h8;
typedef __attribute__((ext_vector_type(2)))  _Float16 h2;
typedef __attribute__((ext_vector_type(8)))  float    v8f;

// ---------------------------------------------------------------------------
// WMMA helper (CDNA5 16x16x32 f16, f32 accumulate)
// ---------------------------------------------------------------------------
__device__ __forceinline__ v8f wmma_f16(v16h a, v16h b, v8f c) {
  return __builtin_amdgcn_wmma_f32_16x16x32_f16(false, a, false, b, (short)0, c,
                                                false, false);
}

// A-matrix 16x32 f16 fragment from a row-major row pointer.
// lane L (m=L%16, hh=L/16): half i -> K = kbase + 16*hh + i (contiguous)
__device__ __forceinline__ v16h load_a_frag(const _Float16* rowptr, int kbase, int hh) {
  union { v16h v; h8 p[2]; } u;
  u.p[0] = *(const h8*)(rowptr + kbase + 16 * hh);
  u.p[1] = *(const h8*)(rowptr + kbase + 16 * hh + 8);
  return u.v;
}

// B-matrix 32x16 fragment where K is contiguous at rowbase (lane's column).
// lane L (n=L%16, hh=L/16): half i -> K = 16*hh + i  => two contiguous b128.
__device__ __forceinline__ v16h load_b_frag_contig(const _Float16* rowbase, int hh) {
  union { v16h v; h8 p[2]; } u;
  u.p[0] = *(const h8*)(rowbase + 16 * hh);
  u.p[1] = *(const h8*)(rowbase + 16 * hh + 8);
  return u.v;
}

// ---------------------------------------------------------------------------
// LayerNorm (f32 in) -> normalized f16 out
// ---------------------------------------------------------------------------
__global__ void __launch_bounds__(256)
ln_f16_kernel(const float* __restrict__ x,
              const float* __restrict__ gamma,
              const float* __restrict__ beta,
              _Float16* __restrict__ xn) {
  const int row = blockIdx.x;
  const float* xr = x + (size_t)row * DIM;
  float vals[4];
  float s = 0.f, s2 = 0.f;
#pragma unroll
  for (int i = 0; i < 4; ++i) {
    float v = xr[threadIdx.x + i * 256];
    vals[i] = v;
    s += v;
    s2 += v * v;
  }
#pragma unroll
  for (int m = 16; m >= 1; m >>= 1) {
    s  += __shfl_xor(s,  m, 32);
    s2 += __shfl_xor(s2, m, 32);
  }
  __shared__ float redS[8], redS2[8];
  const int wv = threadIdx.x >> 5;
  if ((threadIdx.x & 31) == 0) { redS[wv] = s; redS2[wv] = s2; }
  __syncthreads();
  s = 0.f; s2 = 0.f;
#pragma unroll
  for (int i = 0; i < 8; ++i) { s += redS[i]; s2 += redS2[i]; }
  const float mu  = s * (1.0f / DIM);
  const float var = s2 * (1.0f / DIM) - mu * mu;
  const float inv = rsqrtf(var + LN_EPS);
#pragma unroll
  for (int i = 0; i < 4; ++i) {
    const int c = threadIdx.x + i * 256;
    xn[(size_t)row * DIM + c] = (_Float16)((vals[i] - mu) * inv * gamma[c] + beta[c]);
  }
}

// ---------------------------------------------------------------------------
// f32 -> f16 conversion
// ---------------------------------------------------------------------------
__global__ void __launch_bounds__(256)
cvt_f16_kernel(const float* __restrict__ in, _Float16* __restrict__ out, int nElem) {
  const int i = blockIdx.x * blockDim.x + threadIdx.x;
  if (i < nElem) out[i] = (_Float16)in[i];
}

// ---------------------------------------------------------------------------
// f16 GEMM: block = 4 waves; wave computes a 32M x 32N macro-tile (4 WMMAs);
// block tile = 32M x 128N. B chunk staged TRANSPOSED in LDS so every B
// fragment is two ds_load_b128 (K contiguous per column).
// ---------------------------------------------------------------------------
template <bool F16OUT>
__global__ void __launch_bounds__(128)
gemm_wmma_kernel(const _Float16* __restrict__ A, int lda,
                 const _Float16* __restrict__ B, int ldb,
                 void* __restrict__ Cptr, int ldc, int K) {
  // ldsBt[col][kk], col stride 40 halves = 80B (16B aligned rows)
  __shared__ alignas(16) _Float16 ldsBt[128 * 40];
  const int lane = threadIdx.x & 31;
  const int wv   = threadIdx.x >> 5;
  const int hh   = lane >> 4;
  const int ln   = lane & 15;
  const int mblk = blockIdx.y * 32;
  const int nblk = blockIdx.x * 128;
  const _Float16* arow0 = A + (size_t)(mblk + ln) * lda;        // tile rows 0..15
  const _Float16* arow1 = arow0 + (size_t)16 * lda;             // tile rows 16..31
  // staging: thread -> kk rows {2sp, 2sp+1}, cols sc..sc+15
  const int sp = threadIdx.x & 15;
  const int sc = (threadIdx.x >> 4) * 16;
  v8f acc00 = {}, acc01 = {}, acc10 = {}, acc11 = {};

  for (int k0 = 0; k0 < K; k0 += 32) {
    __syncthreads();
    {
      const _Float16* b0 = B + (size_t)(k0 + 2 * sp) * ldb + nblk + sc;
      const _Float16* b1 = b0 + ldb;
      const h8 r0a = *(const h8*)b0, r0b = *(const h8*)(b0 + 8);
      const h8 r1a = *(const h8*)b1, r1b = *(const h8*)(b1 + 8);
      if (k0 + 32 < K) __builtin_prefetch(b0 + (size_t)32 * ldb, 0, 0);
#pragma unroll
      for (int i = 0; i < 8; ++i) {
        h2 t; t[0] = r0a[i]; t[1] = r1a[i];
        *(h2*)&ldsBt[(sc + i) * 40 + 2 * sp] = t;
        h2 u; u[0] = r0b[i]; u[1] = r1b[i];
        *(h2*)&ldsBt[(sc + 8 + i) * 40 + 2 * sp] = u;
      }
    }
    __syncthreads();
    const v16h a0 = load_a_frag(arow0, k0, hh);
    const v16h a1 = load_a_frag(arow1, k0, hh);
    v16h bf0, bf1;
    {
      union { v16h v; h8 p[2]; } u0, u1;
      const _Float16* rb0 = &ldsBt[(wv * 32 + ln) * 40 + 16 * hh];
      const _Float16* rb1 = &ldsBt[(wv * 32 + 16 + ln) * 40 + 16 * hh];
      u0.p[0] = *(const h8*)rb0; u0.p[1] = *(const h8*)(rb0 + 8);
      u1.p[0] = *(const h8*)rb1; u1.p[1] = *(const h8*)(rb1 + 8);
      bf0 = u0.v; bf1 = u1.v;
    }
    acc00 = wmma_f16(a0, bf0, acc00);
    acc01 = wmma_f16(a0, bf1, acc01);
    acc10 = wmma_f16(a1, bf0, acc10);
    acc11 = wmma_f16(a1, bf1, acc11);
  }

  // C layout: vgpr r, half hh -> row r + 8*hh, col = lane%16
  const int c0 = nblk + wv * 32 + ln;
#pragma unroll
  for (int r = 0; r < 8; ++r) {
    const int row0 = mblk + r + 8 * hh;
    const int row1 = row0 + 16;
    if constexpr (F16OUT) {
      _Float16* C = (_Float16*)Cptr;
      C[(size_t)row0 * ldc + c0]      = (_Float16)acc00[r];
      C[(size_t)row0 * ldc + c0 + 16] = (_Float16)acc01[r];
      C[(size_t)row1 * ldc + c0]      = (_Float16)acc10[r];
      C[(size_t)row1 * ldc + c0 + 16] = (_Float16)acc11[r];
    } else {
      float* C = (float*)Cptr;
      C[(size_t)row0 * ldc + c0]      = acc00[r];
      C[(size_t)row0 * ldc + c0 + 16] = acc01[r];
      C[(size_t)row1 * ldc + c0]      = acc10[r];
      C[(size_t)row1 * ldc + c0 + 16] = acc11[r];
    }
  }
}

// ---------------------------------------------------------------------------
// Block-cooperative flash attention: block = 4 waves = one (b, head, 64-query
// block). V tile (32 keys x 64 d) staged transposed in LDS once per key tile,
// shared by all 4 waves. K fragments straight from global (d contiguous).
// ---------------------------------------------------------------------------
__global__ void __launch_bounds__(128)
flash_attn_kernel(const _Float16* __restrict__ qkv,
                  const float* __restrict__ bias,
                  _Float16* __restrict__ out,
                  int n) {
  __shared__ alignas(16) _Float16 ldsVt[64 * 40];      // [d][key], stride 40
  __shared__ alignas(16) _Float16 ldsP[4][16 * 40];    // per-wave P tile

  const int lane = threadIdx.x & 31;
  const int wv   = threadIdx.x >> 5;
  const int hh   = lane >> 4;
  const int ln   = lane & 15;
  const int nqb  = n >> 6;                 // 64-query blocks per (b, h)
  const int bi   = blockIdx.x / (HEADS * nqb);
  const int rem  = blockIdx.x % (HEADS * nqb);
  const int hd   = rem / nqb;
  const int qb   = (rem % nqb) * 64;
  const int q0   = qb + wv * 16;           // this wave's query tile

  const float scale = 1.0f / sqrtf((float)DIM_HEAD);
  const size_t rstride = 3 * (size_t)INNER;
  const _Float16* qrow  = qkv + ((size_t)bi * n + q0 + ln) * rstride + hd * DIM_HEAD;
  const _Float16* kbase = qkv + (size_t)bi * n * rstride + INNER + hd * DIM_HEAD;
  const _Float16* vbase = qkv + (size_t)bi * n * rstride + 2 * INNER + hd * DIM_HEAD;
  const float* brow = bias + ((size_t)hd * n + q0) * n;

  const v16h qa0 = load_a_frag(qrow, 0, hh);
  const v16h qa1 = load_a_frag(qrow, 32, hh);

  v8f o[4] = {v8f{}, v8f{}, v8f{}, v8f{}};
  float rm[8], rs[8];
#pragma unroll
  for (int r = 0; r < 8; ++r) { rm[r] = -3.0e38f; rs[r] = 0.0f; }
  _Float16* P = ldsP[wv];

  // staging assignment: key pair {2sp, 2sp+1}, d cols sd..sd+7
  const int sp = threadIdx.x & 15;
  const int sd = (threadIdx.x >> 4) * 8;
  const int kend = qb + 64;                // block-level causal bound

  for (int jb = 0; jb < kend; jb += 32) {
    __syncthreads();
    {
      const _Float16* v0 = vbase + (size_t)(jb + 2 * sp) * rstride + sd;
      const _Float16* v1 = v0 + rstride;
      const h8 a0 = *(const h8*)v0;
      const h8 a1 = *(const h8*)v1;
#pragma unroll
      for (int i = 0; i < 8; ++i) {
        h2 t; t[0] = a0[i]; t[1] = a1[i];
        *(h2*)&ldsVt[(sd + i) * 40 + 2 * sp] = t;
      }
    }
    __syncthreads();

    if (jb < q0 + 16) {                    // wave-uniform: EXEC stays full
      // S = Q @ K^T for 32 keys (4 WMMAs)
      const _Float16* kr0 = kbase + (size_t)(jb + ln) * rstride;
      const _Float16* kr1 = kr0 + 16 * rstride;
      v8f s0 = {}, s1 = {};
      s0 = wmma_f16(qa0, load_b_frag_contig(kr0, hh), s0);
      s0 = wmma_f16(qa1, load_b_frag_contig(kr0 + 32, hh), s0);
      s1 = wmma_f16(qa0, load_b_frag_contig(kr1, hh), s1);
      s1 = wmma_f16(qa1, load_b_frag_contig(kr1 + 32, hh), s1);

      // scale + bias + causal mask + online softmax
#pragma unroll
      for (int r = 0; r < 8; ++r) {
        const int row = q0 + r + 8 * hh;
        const int k0i = jb + ln, k1i = k0i + 16;
        const float b0 = brow[(size_t)(r + 8 * hh) * n + k0i];
        const float b1 = brow[(size_t)(r + 8 * hh) * n + k1i];
        float v0 = (k0i <= row) ? (s0[r] * scale + b0) : -3.0e38f;
        float v1 = (k1i <= row) ? (s1[r] * scale + b1) : -3.0e38f;
        float t = fmaxf(v0, v1);
#pragma unroll
        for (int m = 8; m >= 1; m >>= 1) t = fmaxf(t, __shfl_xor(t, m, 16));
        const float nm   = fmaxf(rm[r], t);
        const float corr = __expf(rm[r] - nm);
        const float p0 = __expf(v0 - nm);
        const float p1 = __expf(v1 - nm);
        float ps = p0 + p1;
#pragma unroll
        for (int m = 8; m >= 1; m >>= 1) ps += __shfl_xor(ps, m, 16);
        rs[r] = rs[r] * corr + ps;
        rm[r] = nm;
        o[0][r] *= corr; o[1][r] *= corr; o[2][r] *= corr; o[3][r] *= corr;
        P[(r + 8 * hh) * 40 + ln]      = (_Float16)p0;
        P[(r + 8 * hh) * 40 + 16 + ln] = (_Float16)p1;
      }
      asm volatile("s_wait_dscnt 0x0" ::: "memory");

      // P as A fragment: two contiguous b128 reads (stride 40 => 16B aligned)
      v16h pa;
      {
        union { v16h v; h8 p[2]; } u;
        u.p[0] = *(const h8*)&P[ln * 40 + 16 * hh];
        u.p[1] = *(const h8*)&P[ln * 40 + 16 * hh + 8];
        pa = u.v;
      }
      // O += P @ V : V fragments from transposed LDS, 2 b128 each
#pragma unroll
      for (int c = 0; c < 4; ++c) {
        union { v16h v; h8 p[2]; } u;
        const _Float16* vrb = &ldsVt[(c * 16 + ln) * 40 + 16 * hh];
        u.p[0] = *(const h8*)vrb;
        u.p[1] = *(const h8*)(vrb + 8);
        o[c] = wmma_f16(pa, u.v, o[c]);
      }
    }
  }

  // Normalize (reciprocal instead of 32 divides) and store [b, n, INNER] f16
  float rinv[8];
#pragma unroll
  for (int r = 0; r < 8; ++r) rinv[r] = 1.0f / rs[r];
#pragma unroll
  for (int c = 0; c < 4; ++c)
#pragma unroll
    for (int r = 0; r < 8; ++r) {
      const float oval = o[c][r] * rinv[r];
      out[((size_t)bi * n + q0 + r + 8 * hh) * INNER + hd * DIM_HEAD + c * 16 + ln] =
          (_Float16)oval;
    }
}

// ---------------------------------------------------------------------------
// Host-side launch
// ---------------------------------------------------------------------------
extern "C" void kernel_launch(void* const* d_in, const int* in_sizes, int n_in,
                              void* d_out, int out_size, void* d_ws, size_t ws_size,
                              hipStream_t stream) {
  const float* x         = (const float*)d_in[0];
  const float* attn_bias = (const float*)d_in[1];
  const float* ln_gamma  = (const float*)d_in[2];
  const float* ln_beta   = (const float*)d_in[3];
  const float* w_qkv     = (const float*)d_in[4];
  const float* w_out     = (const float*)d_in[5];

  const int bn = in_sizes[0] / DIM;                                   // b*n
  const int n  = (int)(std::sqrt((double)in_sizes[1] / HEADS) + 0.5); // seq len
  const int b  = bn / n;

  _Float16* xn     = (_Float16*)d_ws;                     // bn * DIM
  _Float16* wqkv16 = xn + (size_t)bn * DIM;               // DIM * 3*INNER
  _Float16* wout16 = wqkv16 + (size_t)DIM * 3 * INNER;    // INNER * DIM
  _Float16* qkv    = wout16 + (size_t)INNER * DIM;        // bn * 3*INNER
  _Float16* aout   = qkv + (size_t)bn * 3 * INNER;        // bn * INNER

  ln_f16_kernel<<<bn, 256, 0, stream>>>(x, ln_gamma, ln_beta, xn);

  {
    const int nq = DIM * 3 * INNER;
    cvt_f16_kernel<<<(nq + 255) / 256, 256, 0, stream>>>(w_qkv, wqkv16, nq);
    const int no = INNER * DIM;
    cvt_f16_kernel<<<(no + 255) / 256, 256, 0, stream>>>(w_out, wout16, no);
  }

  {
    dim3 grid((3 * INNER) / 128, bn / 32);
    gemm_wmma_kernel<true><<<grid, 128, 0, stream>>>(xn, DIM, wqkv16, 3 * INNER,
                                                     (void*)qkv, 3 * INNER, DIM);
  }

  {
    const int blocks = b * HEADS * (n / 64);
    flash_attn_kernel<<<blocks, 128, 0, stream>>>(qkv, attn_bias, aout, n);
  }

  {
    dim3 grid(DIM / 128, bn / 32);
    gemm_wmma_kernel<false><<<grid, 128, 0, stream>>>(aout, INNER, wout16, DIM,
                                                      d_out, DIM, INNER);
  }
}
